// CRF_56599079027192
// MI455X (gfx1250) — matrix operand order repaired and checked by reading, not compiled, not run on hardware
//
#include <hip/hip_runtime.h>
#include <hip/hip_bf16.h>

#define NUM_TAGS 17
#define BDIM 2048
#define SDIM 512

typedef _Float16 v16h __attribute__((ext_vector_type(16)));
typedef _Float16 v2h  __attribute__((ext_vector_type(2)));
typedef float    v8f  __attribute__((ext_vector_type(8)));
typedef unsigned int v8u __attribute__((ext_vector_type(8)));

#define LOG2E 1.44269504088896340736f
#define LN2F  0.69314718055994530942f

// Swap data between lane and lane^16 (SWAPX16 via ds_swizzle, wave32).
__device__ __forceinline__ float swap16f(float x) {
    return __int_as_float(__builtin_amdgcn_ds_swizzle(__float_as_int(x), 0x401f));
}

// Pack two f32 into an f16x2 dword (lowers to v_cvt_pk_f16_f32).
__device__ __forceinline__ unsigned pkh2(float x, float y) {
    v2h t;
    t[0] = (_Float16)x;
    t[1] = (_Float16)y;
    return __builtin_bit_cast(unsigned, t);
}

// ---------------------------------------------------------------------------
// Forward algorithm (log-denominator). One wave = 16 batch elements.
// alpha lives in the WMMA C/D layout: lane (half,col) holds tags
// [half*8 .. half*8+7] of batch element (tile*16+col), plus tag 16 replicated.
// Per step (log-sum-exp as scaled matmul):
//   m   = rowmax(alpha); p = exp2((alpha-m)*log2e)        (VALU + ds_swizzle)
//   sum = p (32x16 f16, K zero-padded) x exp(trans)^T     (1x v_wmma_f32_16x16x32_f16)
//   alpha = logit + m + log2(sum)*ln2                     (VALU, native v_log)
// Output tag 16 handled by a VALU rank-1 path; input tag 16 is WMMA K=16.
// ---------------------------------------------------------------------------
__global__ __launch_bounds__(256) void crf_forward_kernel(
    const float* __restrict__ inp,     // (B,S,T)
    const int*   __restrict__ mask,    // (B,S)
    const float* __restrict__ trans,   // (T,T)
    const float* __restrict__ startT,  // (T)
    const float* __restrict__ endT,    // (T)
    float* __restrict__ out)
{
    const int lane  = threadIdx.x & 31;
    const int wave  = (blockIdx.x * blockDim.x + threadIdx.x) >> 5;
    const int half  = lane >> 4;
    const int col   = lane & 15;
    const int bb    = wave * 16 + col;       // batch element owned by this lane
    const int tbase = half * 8;              // first tag row held by this lane

    const unsigned hmask = half ? 0u : 0xffffffffu;   // zero half1's B rows

    // Loop-invariant A matrix: A[m][k] = exp(trans[k][m]), m = col.
    // f16 A layout (16x32): half0 h[0..7]=K0..7, h[8..15]=K16..23;
    //                       half1 h[0..7]=K8..15, h[8..15]=K24..31.
    // Real K stops at 16 (h[8] of half0); the rest is zero padding.
    v16h EA;
#pragma unroll
    for (int i = 0; i < 16; ++i) EA[i] = (_Float16)0.f;
#pragma unroll
    for (int i = 0; i < 8; ++i) {
        const int k = half ? (8 + i) : i;
        EA[i] = (_Float16)__builtin_amdgcn_exp2f(trans[k * NUM_TAGS + col] * LOG2E);
    }
    {   // K = 16 lives in half0's h[8]
        const float e16 = __builtin_amdgcn_exp2f(trans[16 * NUM_TAGS + col] * LOG2E);
        EA[8] = half ? (_Float16)0.f : (_Float16)e16;
    }

    // Constants for the output-tag-16 rank-1 path: E[tbase+r][16], E[16][16].
    float e16col[8];
#pragma unroll
    for (int r = 0; r < 8; ++r)
        e16col[r] = __builtin_amdgcn_exp2f(trans[(tbase + r) * NUM_TAGS + 16] * LOG2E);
    const float e16_16 = __builtin_amdgcn_exp2f(trans[16 * NUM_TAGS + 16] * LOG2E);

    // alpha0 = inputs[:,0,:] + start_transitions
    const float* rowp = inp + (size_t)bb * SDIM * NUM_TAGS;
    float a[8], a16;
#pragma unroll
    for (int r = 0; r < 8; ++r) a[r] = rowp[tbase + r] + startT[tbase + r];
    a16 = rowp[16] + startT[16];

    for (int s = 1; s < SDIM; ++s) {
        rowp += NUM_TAGS;
        __builtin_prefetch(rowp + 8 * NUM_TAGS, 0, 1);   // ~8 steps ahead

        float lg[8];
#pragma unroll
        for (int r = 0; r < 8; ++r) lg[r] = rowp[tbase + r];
        const float lg16 = rowp[16];
        const float mf   = (float)mask[(size_t)bb * SDIM + s];

        // row max over all 17 tags (cross-half via swizzle; a16 replicated)
        float mx = a[0];
#pragma unroll
        for (int r = 1; r < 8; ++r) mx = fmaxf(mx, a[r]);
        mx = fmaxf(mx, swap16f(mx));
        mx = fmaxf(mx, a16);

        // p = exp2((alpha - mx) * log2e)  — raw v_exp_f32, args <= 0 so safe
        const float nm = -mx * LOG2E;
        float p[8];
#pragma unroll
        for (int r = 0; r < 8; ++r)
            p[r] = __builtin_amdgcn_exp2f(fmaf(a[r], LOG2E, nm));
        const float p16 = __builtin_amdgcn_exp2f(fmaf(a16, LOG2E, nm));

        // other half's p (half0 receives tags 8..15, half1 receives 0..7)
        float q[8];
#pragma unroll
        for (int r = 0; r < 8; ++r) q[r] = swap16f(p[r]);

        // B matrix (32x16 f16): half0 lane h[0..7] = K0..7 = p0..7, h[8] = K16 = p16;
        // half1 lane h[0..7] = K8..15 = p0..7 (its own tags 8..15); rest zero.
        v8u ub;
        ub[0] = pkh2(p[0], p[1]);
        ub[1] = pkh2(p[2], p[3]);
        ub[2] = pkh2(p[4], p[5]);
        ub[3] = pkh2(p[6], p[7]);
        ub[4] = pkh2(p16, 0.f) & hmask;
        ub[5] = 0u;
        ub[6] = 0u;
        ub[7] = 0u;
        const v16h PB = __builtin_bit_cast(v16h, ub);

        v8f acc = {};
        acc = __builtin_amdgcn_wmma_f32_16x16x32_f16(
                  false, EA, false, PB, (short)0, acc, false, false);

        // output tag 16: new16 = sum_i p_i * E[i][16]  (independent of acc —
        // fills the WMMA->VALU hazard window)
        float part = 0.f;
#pragma unroll
        for (int r = 0; r < 8; ++r) part = fmaf(p[r], e16col[r], part);
        const float tot16 = fmaf(p16, e16_16, part + swap16f(part));

        // alpha = logit + mx + log2(sum)*ln2, blended with mask:
        // a += mf * (na - a)
#pragma unroll
        for (int r = 0; r < 8; ++r) {
            const float na = fmaf(__builtin_amdgcn_logf(acc[r]), LN2F, lg[r] + mx);
            a[r] = fmaf(mf, na - a[r], a[r]);
        }
        {
            const float na = fmaf(__builtin_amdgcn_logf(tot16), LN2F, lg16 + mx);
            a16 = fmaf(mf, na - a16, a16);
        }
    }

    // log_denominator = logsumexp(alpha + end_transitions)
    float v[8];
#pragma unroll
    for (int r = 0; r < 8; ++r) v[r] = a[r] + endT[tbase + r];
    const float v16 = a16 + endT[16];
    float mx = v[0];
#pragma unroll
    for (int r = 1; r < 8; ++r) mx = fmaxf(mx, v[r]);
    mx = fmaxf(mx, swap16f(mx));
    mx = fmaxf(mx, v16);
    const float nm = -mx * LOG2E;
    float sum = 0.f;
#pragma unroll
    for (int r = 0; r < 8; ++r)
        sum += __builtin_amdgcn_exp2f(fmaf(v[r], LOG2E, nm));
    sum += swap16f(sum);
    sum += __builtin_amdgcn_exp2f(fmaf(v16, LOG2E, nm));
    const float lse = fmaf(__builtin_amdgcn_logf(sum), LN2F, mx);

    if (half == 0) atomicAdd(out, -lse);   // one contribution per batch element
}

// ---------------------------------------------------------------------------
// Numerator score: one thread per batch element (pure gather/accumulate).
// ---------------------------------------------------------------------------
__global__ __launch_bounds__(256) void crf_score_kernel(
    const float* __restrict__ inp,
    const int*   __restrict__ tags,
    const int*   __restrict__ mask,
    const float* __restrict__ trans,
    const float* __restrict__ startT,
    const float* __restrict__ endT,
    float* __restrict__ out)
{
    const int b = blockIdx.x * blockDim.x + threadIdx.x;
    if (b >= BDIM) return;
    const int*   tg = tags + (size_t)b * SDIM;
    const int*   mk = mask + (size_t)b * SDIM;
    const float* ip = inp  + (size_t)b * SDIM * NUM_TAGS;

    float score = startT[tg[0]];
    int cnt = 0;
    for (int s = 0; s < SDIM; ++s) cnt += mk[s];

    for (int s = 0; s + 1 < SDIM; ++s) {
        const int cur = tg[s], nxt = tg[s + 1];
        score += trans[cur * NUM_TAGS + nxt] * (float)mk[s + 1]
               + ip[(size_t)s * NUM_TAGS + cur] * (float)mk[s];
    }
    const int last_tag = tg[cnt - 1];
    score += endT[last_tag]
           + ip[(size_t)(SDIM - 1) * NUM_TAGS + last_tag] * (float)mk[SDIM - 1];

    atomicAdd(out, score);
}

__global__ void crf_zero_kernel(float* out) { out[0] = 0.f; }

// ---------------------------------------------------------------------------
extern "C" void kernel_launch(void* const* d_in, const int* in_sizes, int n_in,
                              void* d_out, int out_size, void* d_ws, size_t ws_size,
                              hipStream_t stream) {
    const float* inputs = (const float*)d_in[0];   // (2048,512,17) f32
    const int*   tags   = (const int*)  d_in[1];   // (2048,512)
    const int*   mask   = (const int*)  d_in[2];   // (2048,512)
    const float* trans  = (const float*)d_in[3];   // (17,17)
    const float* startT = (const float*)d_in[4];   // (17)
    const float* endT   = (const float*)d_in[5];   // (17)
    float* out = (float*)d_out;

    crf_zero_kernel<<<1, 1, 0, stream>>>(out);

    // Forward: 2048/16 = 128 waves * 32 lanes = 4096 threads.
    const int fwd_threads = (BDIM / 16) * 32;
    crf_forward_kernel<<<fwd_threads / 256, 256, 0, stream>>>(
        inputs, mask, trans, startT, endT, out);

    crf_score_kernel<<<(BDIM + 255) / 256, 256, 0, stream>>>(
        inputs, tags, mask, trans, startT, endT, out);
}
